// PersonaAligner_53240414601682
// MI455X (gfx1250) — matrix operand order repaired and checked by reading, not compiled, not run on hardware
//
#include <hip/hip_runtime.h>
#include <hip/hip_bf16.h>
#include <math.h>

// ---------------- problem constants ----------------
#define BB    64      // batch
#define SS    128     // seq len
#define DD    512     // embed dim
#define HH    1024    // hidden
#define G4H   4096    // 4*H
#define KTOT  1536    // D + H (fused GEMM K)
#define CMAX  64
#define WW    24

typedef __bf16 bf16_t;
typedef __attribute__((ext_vector_type(16))) __bf16 v16bf;
typedef __attribute__((ext_vector_type(8)))  float  v8f;

// ---------------- workspace layout (bytes) ----------------
// Wcat  : bf16 [4096][1536]   = 12,582,912 B
// bias  : f32  [4096]         =     16,384 B
// hn    : f32  [64][1024]     =    262,144 B
// q     : f32  [64][512]      =    131,072 B
#define WS_WCAT_OFF 0
#define WS_BIAS_OFF (WS_WCAT_OFF + G4H * KTOT * 2)
#define WS_HN_OFF   (WS_BIAS_OFF + G4H * 4)
#define WS_Q_OFF    (WS_HN_OFF   + BB * HH * 4)

#define LOG2E 1.4426950408889634f

// branch-free activations: v_exp_f32 (base-2 TRANS op) + v_rcp_f32
__device__ __forceinline__ float fast_sigmoid(float x) {
    return __builtin_amdgcn_rcpf(1.0f + __builtin_amdgcn_exp2f(-x * LOG2E));
}
__device__ __forceinline__ float fast_tanh(float x) {
    // tanh(x) = 1 - 2/(exp(2x)+1); saturates correctly at +/-inf, no branches
    return 1.0f - 2.0f * __builtin_amdgcn_rcpf(
                      1.0f + __builtin_amdgcn_exp2f(x * (2.0f * LOG2E)));
}

// ============================================================
// K0: build fused bf16 weight [w_ih | w_hh] and bias = b_ih+b_hh
// ============================================================
__global__ void prep_weights(const float* __restrict__ w_ih,
                             const float* __restrict__ w_hh,
                             const float* __restrict__ b_ih,
                             const float* __restrict__ b_hh,
                             bf16_t* __restrict__ Wcat,
                             float*  __restrict__ bias) {
    const int stride = gridDim.x * blockDim.x;
    for (int i = blockIdx.x * blockDim.x + threadIdx.x;
         i < G4H * KTOT; i += stride) {
        const int g = i / KTOT;
        const int k = i - g * KTOT;
        const float v = (k < DD) ? w_ih[g * DD + k] : w_hh[g * HH + (k - DD)];
        Wcat[i] = (bf16_t)v;
    }
    const int t = blockIdx.x * blockDim.x + threadIdx.x;
    if (t < G4H) bias[t] = b_ih[t] + b_hh[t];
}

// ============================================================
// K1: LSTM over time with packed-sequence masking.
//   4 workgroups x 512 threads; each WG owns 16 batch rows.
//   gates[16,4096] = [emb(x_t) | h_{t-1}] (16x1536 bf16) @ Wcat^T  + bias
//   via v_wmma_f32_16x16x32_bf16, f32 accumulation.
//   Runs only t < max(len) of the WG's rows (uniform early exit).
// ============================================================
__global__ __launch_bounds__(512)
void lstm_kernel(const int*   __restrict__ src_seqs,   // [B,S]
                 const int*   __restrict__ src_len,    // [B]
                 const float* __restrict__ emb,        // [V,D]
                 const bf16_t* __restrict__ Wcat,      // [4096,1536]
                 const float* __restrict__ bias,       // [4096]
                 float* __restrict__ hn_out)           // [B,H]
{
    __shared__ bf16_t a_lds[16 * KTOT];   // rows: [x(512) | h(1024)]  48 KB
    __shared__ int    tok_lds[16];
    __shared__ int    len_lds[16];

    const int tid    = threadIdx.x;
    const int wave   = tid >> 5;          // 0..15
    const int lane   = tid & 31;
    const int hs     = lane >> 4;         // half-select
    const int ln     = lane & 15;
    const int batch0 = blockIdx.x * 16;

    if (tid < 16) len_lds[tid] = src_len[batch0 + tid];
    // zero the h region of the A operand (h_{-1} = 0)
    for (int i = tid; i < 16 * HH; i += 512) {
        const int r = i >> 10;
        const int c = i & (HH - 1);
        a_lds[r * KTOT + DD + c] = (bf16_t)0.0f;
    }
    __syncthreads();

    int lenj[8];
#pragma unroll
    for (int e = 0; e < 8; ++e) lenj[e] = len_lds[hs * 8 + e];

    // WG-uniform trip count: beyond max(len) every update is a no-op
    int tmax = 1;
#pragma unroll
    for (int r = 0; r < 16; ++r) tmax = max(tmax, len_lds[r]);

    // bias fragments are time-invariant: preload
    float bv[4][4];   // [u][g]
#pragma unroll
    for (int u = 0; u < 4; ++u)
#pragma unroll
        for (int g = 0; g < 4; ++g)
            bv[u][g] = bias[g * HH + (wave * 4 + u) * 16 + ln];

    float cc[4][8];   // cell state fragments (4 hidden tiles per wave)
    float hk[4][8];   // length-masked hidden state (f32, kept for h_n)
#pragma unroll
    for (int u = 0; u < 4; ++u)
#pragma unroll
        for (int e = 0; e < 8; ++e) { cc[u][e] = 0.0f; hk[u][e] = 0.0f; }

    for (int t = 0; t < tmax; ++t) {
        if (tid < 16) tok_lds[tid] = src_seqs[(batch0 + tid) * SS + t];
        __syncthreads();   // prev-step reads of a_lds complete; tokens ready

        // gather x_t embeddings -> bf16 A operand rows, 16x512
        for (int i = tid; i < 16 * (DD / 4); i += 512) {
            const int r  = i >> 7;                // DD/4 = 128
            const int c4 = i & 127;
            const float4 v = ((const float4*)(emb + (size_t)tok_lds[r] * DD))[c4];
            bf16_t* dst = &a_lds[r * KTOT + c4 * 4];
            dst[0] = (bf16_t)v.x; dst[1] = (bf16_t)v.y;
            dst[2] = (bf16_t)v.z; dst[3] = (bf16_t)v.w;
        }
        __syncthreads();   // x_t ready, h_{t-1} ready

        // per-step arithmetic update mask (branch-free)
        float selv[8];
#pragma unroll
        for (int e = 0; e < 8; ++e) selv[e] = (t < lenj[e]) ? 1.0f : 0.0f;

#pragma unroll
        for (int u = 0; u < 4; ++u) {
            const int ht = wave * 4 + u;          // hidden tile 0..63
            v8f acc[4];
#pragma unroll
            for (int g = 0; g < 4; ++g)
#pragma unroll
                for (int e = 0; e < 8; ++e) acc[g][e] = bv[u][g];

            for (int kc = 0; kc < KTOT; kc += 32) {
                // A fragment (16-bit A layout: lane<16 K{0..7,16..23})
                const bf16_t* arow = &a_lds[ln * KTOT + kc + hs * 8];
                v16bf a;
#pragma unroll
                for (int e = 0; e < 8; ++e) { a[e] = arow[e]; a[8 + e] = arow[16 + e]; }
#pragma unroll
                for (int g = 0; g < 4; ++g) {
                    // B fragment: lane holds K{hs*8..+7, 16+hs*8..+7} of Wcat row (g0+ln)
                    const bf16_t* wrow =
                        Wcat + (size_t)(g * HH + ht * 16 + ln) * KTOT + kc + hs * 8;
                    v16bf bf;
#pragma unroll
                    for (int e = 0; e < 8; ++e) { bf[e] = wrow[e]; bf[8 + e] = wrow[16 + e]; }
                    acc[g] = __builtin_amdgcn_wmma_f32_16x16x32_bf16(
                        false, a, false, bf, (short)0, acc[g], false, false);
                }
            }
            // elementwise LSTM cell on fragments (row m = hs*8+e), branch-free
#pragma unroll
            for (int e = 0; e < 8; ++e) {
                const float ii = fast_sigmoid(acc[0][e]);
                const float ff = fast_sigmoid(acc[1][e]);
                const float gg = fast_tanh(acc[2][e]);
                const float oo = fast_sigmoid(acc[3][e]);
                const float cn = ff * cc[u][e] + ii * gg;
                const float hv = oo * fast_tanh(cn);
                const float s  = selv[e];
                cc[u][e] = fmaf(s, cn - cc[u][e], cc[u][e]);
                hk[u][e] = fmaf(s, hv - hk[u][e], hk[u][e]);
            }
        }
        __syncthreads();   // every wave done reading h_{t-1}

        // publish masked h_t (bf16) into the A operand's h region
#pragma unroll
        for (int u = 0; u < 4; ++u) {
            const int n0 = (wave * 4 + u) * 16;
#pragma unroll
            for (int e = 0; e < 8; ++e) {
                const int m = hs * 8 + e;
                a_lds[m * KTOT + DD + n0 + ln] = (bf16_t)hk[u][e];
            }
        }
    }

    // write h_n (f32 kept values) to workspace
#pragma unroll
    for (int u = 0; u < 4; ++u) {
        const int n0 = (wave * 4 + u) * 16;
#pragma unroll
        for (int e = 0; e < 8; ++e) {
            const int m = hs * 8 + e;
            hn_out[(size_t)(batch0 + m) * HH + n0 + ln] = hk[u][e];
        }
    }
}

// ============================================================
// K2: q = h_n @ M_w.T + M_b     (64 x 512, K=1024)
// ============================================================
__global__ __launch_bounds__(512)
void q_kernel(const float* __restrict__ hn,
              const float* __restrict__ Mw,   // [D,H]
              const float* __restrict__ Mb,   // [D]
              float* __restrict__ q) {
    const int b = blockIdx.x;
    const int d = threadIdx.x;
    const float* hrow = hn + (size_t)b * HH;
    const float* wrow = Mw + (size_t)d * HH;
    float acc = Mb[d];
    for (int h = 0; h < HH; h += 4) {
        acc += hrow[h + 0] * wrow[h + 0];
        acc += hrow[h + 1] * wrow[h + 1];
        acc += hrow[h + 2] * wrow[h + 2];
        acc += hrow[h + 3] * wrow[h + 3];
    }
    q[(size_t)b * DD + d] = acc;
}

// ============================================================
// K3: fused embedding-bag + dot:
//   logits[b,c] = (sum_{w<len} q_b . emb[tok]) / len  (mean is linear)
//   8 waves per block, one (b,c) pair per wave.
//   c==CMAX -> 0;  c>=ctc_len -> -inf
// ============================================================
__global__ __launch_bounds__(256)
void logits_kernel(const float* __restrict__ q,          // [B,D]
                   const float* __restrict__ emb,        // [V,D]
                   const int*   __restrict__ ctc_tokens, // [B,CMAX,W]
                   const int*   __restrict__ ctc_wlen,   // [B,CMAX]
                   const int*   __restrict__ ctc_len,    // [B]
                   float* __restrict__ out)              // [B,CMAX+1]
{
    const int idx  = blockIdx.x * 8 + (threadIdx.x >> 5); // b*(CMAX+1)+c
    const int lane = threadIdx.x & 31;
    if (idx >= BB * (CMAX + 1)) return;
    const int b = idx / (CMAX + 1);
    const int c = idx - b * (CMAX + 1);

    if (c == CMAX) { if (lane == 0) out[idx] = 0.0f; return; }
    if (c >= ctc_len[b]) { if (lane == 0) out[idx] = -__builtin_inff(); return; }

    const int wl = ctc_wlen[b * CMAX + c];
    float qr[16];
#pragma unroll
    for (int k = 0; k < 16; ++k) qr[k] = q[(size_t)b * DD + k * 32 + lane];

    float acc = 0.0f;
    const int* toks = ctc_tokens + ((size_t)b * CMAX + c) * WW;
    for (int w = 0; w < wl; ++w) {
        const float* er = emb + (size_t)toks[w] * DD;
#pragma unroll
        for (int k = 0; k < 16; ++k) acc += qr[k] * er[k * 32 + lane];
    }
#pragma unroll
    for (int off = 16; off > 0; off >>= 1) acc += __shfl_xor(acc, off, 32);
    if (lane == 0) out[idx] = acc / (float)wl;
}

// ============================================================
extern "C" void kernel_launch(void* const* d_in, const int* in_sizes, int n_in,
                              void* d_out, int out_size, void* d_ws, size_t ws_size,
                              hipStream_t stream) {
    const int*   src_seqs  = (const int*)  d_in[0];
    const int*   src_len   = (const int*)  d_in[1];
    const int*   ctc_tok   = (const int*)  d_in[2];
    const int*   ctc_wlen  = (const int*)  d_in[3];
    const int*   ctc_len   = (const int*)  d_in[4];
    const float* emb       = (const float*)d_in[5];
    const float* w_ih      = (const float*)d_in[6];
    const float* w_hh      = (const float*)d_in[7];
    const float* b_ih      = (const float*)d_in[8];
    const float* b_hh      = (const float*)d_in[9];
    const float* M_w       = (const float*)d_in[10];
    const float* M_b       = (const float*)d_in[11];
    float* out = (float*)d_out;

    char* ws = (char*)d_ws;
    bf16_t* Wcat = (bf16_t*)(ws + WS_WCAT_OFF);
    float*  bias = (float*) (ws + WS_BIAS_OFF);
    float*  hn   = (float*) (ws + WS_HN_OFF);
    float*  q    = (float*) (ws + WS_Q_OFF);

    prep_weights<<<4096, 256, 0, stream>>>(w_ih, w_hh, b_ih, b_hh, Wcat, bias);
    lstm_kernel<<<BB / 16, 512, 0, stream>>>(src_seqs, src_len, emb, Wcat, bias, hn);
    q_kernel<<<BB, DD, 0, stream>>>(hn, M_w, M_b, q);

    const int npairs = BB * (CMAX + 1);
    logits_kernel<<<(npairs + 7) / 8, 256, 0, stream>>>(q, emb, ctc_tok, ctc_wlen,
                                                        ctc_len, out);
}
